// CrossRec_85409719648304
// MI455X (gfx1250) — matrix operand order repaired and checked by reading, not compiled, not run on hardware
//
#include <hip/hip_runtime.h>
#include <hip/hip_bf16.h>

#define N_RNA  4096
#define N_ATAC 8192
#define IN_C   256
#define HID    128
#define NH     2
#define DH     64
#define TOPK   10
#define THR    0.8f

typedef __attribute__((ext_vector_type(16))) __bf16 v16bf;
typedef __attribute__((ext_vector_type(8)))  __bf16 v8bf;
typedef __attribute__((ext_vector_type(8)))  float  v8f;
typedef __attribute__((ext_vector_type(4)))  float  v4f;

// ---------------------------------------------------------------------------
// WMMA operand loaders from pre-converted bf16 arrays (CDNA5 ISA):
//   A 16x32: per lane K = ks+{0..7, 16..23}, ks = 8*(lane>=16)  -> 2x16B loads
//   B 32x16: per lane K = ks+{0..7, 16..23}, ks = 8*(lane>=16)  -> 2x16B loads
// ---------------------------------------------------------------------------
__device__ __forceinline__ v16bf loadA16(const __bf16* __restrict__ p) {
  v8bf a = *(const v8bf*)p;
  v8bf b = *(const v8bf*)(p + 16);
  return __builtin_shufflevector(a, b, 0,1,2,3,4,5,6,7,8,9,10,11,12,13,14,15);
}
__device__ __forceinline__ v16bf loadB16(const __bf16* __restrict__ p) {
  v8bf a = *(const v8bf*)p;
  v8bf b = *(const v8bf*)(p + 16);
  return __builtin_shufflevector(a, b, 0,1,2,3,4,5,6,7,8,9,10,11,12,13,14,15);
}

// hi*hi + hi*lo + lo*hi ~= fp32 product, f32 accumulate
__device__ __forceinline__ v8f wmma3(v16bf ah, v16bf al, v16bf bh, v16bf bl, v8f c) {
  c = __builtin_amdgcn_wmma_f32_16x16x32_bf16(false, ah, false, bh, (short)0, c, false, false);
  c = __builtin_amdgcn_wmma_f32_16x16x32_bf16(false, ah, false, bl, (short)0, c, false, false);
  c = __builtin_amdgcn_wmma_f32_16x16x32_bf16(false, al, false, bh, (short)0, c, false, false);
  return c;
}

__device__ __forceinline__ void split_bf16(float x, __bf16& h, __bf16& l) {
  h = (__bf16)x;
  l = (__bf16)(x - (float)h);
}

// ---------------------------------------------------------------------------
// One-time converters (f32 -> split-bf16 hi/lo pairs)
// ---------------------------------------------------------------------------
__global__ void fconv_hl(const float* __restrict__ in, __bf16* __restrict__ h,
                         __bf16* __restrict__ l, int n) {
  int i = blockIdx.x * 256 + threadIdx.x;
  if (i < n) { __bf16 hh, ll; split_bf16(in[i], hh, ll); h[i] = hh; l[i] = ll; }
}

// W [K][128] f32 -> transposed split-bf16 [128][K] (per-lane-contiguous B loads)
__global__ void wconv_T(const float* __restrict__ W, __bf16* __restrict__ Th,
                        __bf16* __restrict__ Tl, int K) {
  int i = blockIdx.x * 256 + threadIdx.x;
  if (i < K * HID) {
    int k = i >> 7, n = i & (HID - 1);
    __bf16 hh, ll; split_bf16(W[i], hh, ll);
    Th[(size_t)n * K + k] = hh;
    Tl[(size_t)n * K + k] = ll;
  }
}

__global__ void zerof(float* __restrict__ p, int n) {
  int i = blockIdx.x * 256 + threadIdx.x;
  if (i < n) p[i] = 0.f;
}

// ---------------------------------------------------------------------------
// GEMM: Y[N x 128] = X[N x K] @ W^T + bias.  X as split-bf16 [N][K],
// W as transposed split-bf16 [128][ldw] (+offset folded into pointer).
// mode 0: Y f32 (+= if accum);  mode 1: Y written as split-bf16 pair.
// block = 256 thr = 8 waves; wave w -> 16x16 tile at col 16w, rows 16*blockIdx.
// ---------------------------------------------------------------------------
__global__ __launch_bounds__(256)
void gemm_bf16hl(const __bf16* __restrict__ Xh, const __bf16* __restrict__ Xl,
                 const __bf16* __restrict__ Wh, const __bf16* __restrict__ Wl,
                 const float* __restrict__ bias,
                 float* __restrict__ Yf,
                 __bf16* __restrict__ Yh, __bf16* __restrict__ Yl,
                 int K, int ldw, int mode, int accum) {
  const int m0   = blockIdx.x * 16;
  const int n0   = (threadIdx.x >> 5) * 16;
  const int lane = threadIdx.x & 31;
  const int lh   = lane >> 4, l15 = lane & 15;

  const __bf16* xh = Xh + (size_t)(m0 + l15) * K + lh * 8;
  const __bf16* xl = Xl + (size_t)(m0 + l15) * K + lh * 8;
  const __bf16* wh = Wh + (size_t)(n0 + l15) * ldw + lh * 8;
  const __bf16* wl = Wl + (size_t)(n0 + l15) * ldw + lh * 8;

  v8f acc = {};
  for (int kc = 0; kc < K; kc += 32) {
    v16bf ah = loadA16(xh + kc);
    v16bf al = loadA16(xl + kc);
    v16bf bh = loadB16(wh + kc);
    v16bf bl = loadB16(wl + kc);
    acc = wmma3(ah, al, bh, bl, acc);
  }
#pragma unroll
  for (int j = 0; j < 8; ++j) {
    const int m = m0 + j + lh * 8, n = n0 + l15;
    float r = acc[j];
    if (bias) r += bias[n];
    const size_t o = (size_t)m * HID + n;
    if (mode == 0) {
      if (accum) r += Yf[o];
      Yf[o] = r;
    } else {
      __bf16 hh, ll; split_bf16(r, hh, ll);
      Yh[o] = hh; Yl[o] = ll;
    }
  }
}

// ---------------------------------------------------------------------------
// Fused QK^T -> mask -> top-10 -> sigmoid/softmax/threshold.
// block = 256 thr = 8 waves owns (head h, rows r0..r0+15); slabs of 128 cols.
// Double-buffered LDS score slab -> one barrier per iteration.
// Mask (the 128MB stream) applied in the scan stage with coalesced v4f loads.
// ---------------------------------------------------------------------------
__global__ __launch_bounds__(256)
void attn_topk(const __bf16* __restrict__ Qh, const __bf16* __restrict__ Ql,
               const __bf16* __restrict__ Kh, const __bf16* __restrict__ Kl,
               const float* __restrict__ mask,
               float* __restrict__ topw, int* __restrict__ topi) {
  const int blk  = blockIdx.x;           // [0, 512)
  const int h    = blk >> 8;
  const int r0   = (blk & 255) * 16;
  const int wv   = threadIdx.x >> 5;     // 0..7 -> 16-col sub-slab
  const int lane = threadIdx.x & 31;
  const int lh   = lane >> 4, l15 = lane & 15;

  __shared__ float sc[2][16 * 128];
  __shared__ float candv[16][16 * TOPK];
  __shared__ int   candi[16][16 * TOPK];

  // Q tile (16 x 64), two K=32 chunks, resident in registers for all 64 iters
  v16bf ah[2], al[2];
#pragma unroll
  for (int c = 0; c < 2; ++c) {
    const size_t qo = (size_t)(r0 + l15) * HID + h * DH + c * 32 + lh * 8;
    ah[c] = loadA16(Qh + qo);
    al[c] = loadA16(Ql + qo);
  }

  float tv[TOPK]; int ti[TOPK];
#pragma unroll
  for (int j = 0; j < TOPK; ++j) { tv[j] = -1e30f; ti[j] = 0; }
  const int row = threadIdx.x >> 4;        // 0..15
  const int cs  = (threadIdx.x & 15) * 8;  // 0..120

  int p = 0;
  for (int a0 = 0; a0 < N_ATAC; a0 += 128, p ^= 1) {
    const int acol = a0 + wv * 16 + l15;
    v8f acc = {};
#pragma unroll
    for (int c = 0; c < 2; ++c) {
      const size_t ko = (size_t)acol * HID + h * DH + c * 32 + lh * 8;
      v16bf bh = loadB16(Kh + ko);
      v16bf bl = loadB16(Kl + ko);
      acc = wmma3(ah[c], al[c], bh, bl, acc);
    }
#pragma unroll
    for (int j = 0; j < 8; ++j)
      sc[p][(j + lh * 8) * 128 + wv * 16 + l15] = acc[j];
    __syncthreads();

    // scan: 16 threads per row, 8 consecutive cols each; mask from global
    const float* mrow = mask + (size_t)(r0 + row) * N_ATAC + a0 + cs;
    if (a0 + 128 < N_ATAC) __builtin_prefetch(mrow + 128, 0, 1);
    v4f mA = *(const v4f*)mrow;
    v4f mB = *(const v4f*)(mrow + 4);
#pragma unroll
    for (int c = 0; c < 8; ++c) {
      const float mv = (c < 4) ? mA[c] : mB[c - 4];
      float v = sc[p][row * 128 + cs + c] * mv;   // sigmoid monotone: rank logits
      if (v > tv[TOPK - 1]) {
        int gi = a0 + cs + c;
#pragma unroll
        for (int j = 0; j < TOPK; ++j)
          if (v > tv[j]) { float t = tv[j]; tv[j] = v; v = t;
                           int   u = ti[j]; ti[j] = gi; gi = u; }
      }
    }
    // no trailing barrier: next iter writes the other LDS buffer
  }

  // merge 16 partial lists per row
  const int slot = threadIdx.x & 15;
#pragma unroll
  for (int j = 0; j < TOPK; ++j) {
    candv[row][slot * TOPK + j] = tv[j];
    candi[row][slot * TOPK + j] = ti[j];
  }
  __syncthreads();

  if (threadIdx.x < 16) {
    const int r = threadIdx.x;
    float fv[TOPK]; int fi[TOPK];
#pragma unroll
    for (int j = 0; j < TOPK; ++j) { fv[j] = -1e30f; fi[j] = 0; }
    for (int c = 0; c < 16 * TOPK; ++c) {
      float v = candv[r][c];
      if (v > fv[TOPK - 1]) {
        int gi = candi[r][c];
#pragma unroll
        for (int j = 0; j < TOPK; ++j)
          if (v > fv[j]) { float t = fv[j]; fv[j] = v; v = t;
                           int   u = fi[j]; fi[j] = gi; gi = u; }
      }
    }
    float sg[TOPK], mx = -1e30f;
#pragma unroll
    for (int j = 0; j < TOPK; ++j) { sg[j] = 1.f / (1.f + __expf(-fv[j])); mx = fmaxf(mx, sg[j]); }
    float s = 0.f, e[TOPK];
#pragma unroll
    for (int j = 0; j < TOPK; ++j) { e[j] = __expf(sg[j] - mx); s += e[j]; }
    const float inv = 1.f / s;
    const size_t base = ((size_t)h * N_RNA + (r0 + r)) * TOPK;
#pragma unroll
    for (int j = 0; j < TOPK; ++j) {
      topw[base + j] = (sg[j] > THR) ? e[j] * inv : 0.f;
      topi[base + j] = fi[j];
    }
  }
}

// rna_agg[r,d] = sum_j w * v_a[idx, d]; written directly as split-bf16
__global__ __launch_bounds__(128)
void rna_gather(const float* __restrict__ Va, const float* __restrict__ topw,
                const int* __restrict__ topi,
                __bf16* __restrict__ Rh, __bf16* __restrict__ Rl) {
  const int r = blockIdx.x;
  const int d = threadIdx.x;           // 0..127
  const int h = d >> 6;
  const size_t base = ((size_t)h * N_RNA + r) * TOPK;
  float acc = 0.f;
#pragma unroll
  for (int j = 0; j < TOPK; ++j)
    acc += topw[base + j] * Va[(size_t)topi[base + j] * HID + d];
  __bf16 hh, ll; split_bf16(acc, hh, ll);
  const size_t o = (size_t)r * HID + d;
  Rh[o] = hh; Rl[o] = ll;
}

// atac_agg[a, h*64+d] += w * v_r[r, h*64+d] via native global f32 atomics
__global__ __launch_bounds__(64)
void atac_scatter(const float* __restrict__ Vr, const float* __restrict__ topw,
                  const int* __restrict__ topi, float* __restrict__ out) {
  const int blk = blockIdx.x;          // h*4096 + r
  const int h = blk >> 12, r = blk & (N_RNA - 1);
  const int d = threadIdx.x;           // 0..63
  const size_t base = ((size_t)h * N_RNA + r) * TOPK;
  const float v = Vr[(size_t)r * HID + (h << 6) + d];
#pragma unroll
  for (int j = 0; j < TOPK; ++j) {
    const float w = topw[base + j];
    if (w != 0.f) {
      float* p = &out[(size_t)topi[base + j] * HID + (h << 6) + d];
      (void)__hip_atomic_fetch_add(p, w * v, __ATOMIC_RELAXED, __HIP_MEMORY_SCOPE_AGENT);
    }
  }
}

// ---------------------------------------------------------------------------
extern "C" void kernel_launch(void* const* d_in, const int* in_sizes, int n_in,
                              void* d_out, int out_size, void* d_ws, size_t ws_size,
                              hipStream_t stream) {
  const float* x_rna = (const float*)d_in[0];
  const float* x_atac = (const float*)d_in[1];
  const float* cmask = (const float*)d_in[2];
  const float* W_[11]; const float* b_[11];
  // Wq,Wk,Wvr,Wva,Wor,Woa,Wsr,Wsa,Wdr,Wda with their biases (input dict order)
  W_[0] = (const float*)d_in[3];  b_[0] = (const float*)d_in[4];   // q
  W_[1] = (const float*)d_in[5];  b_[1] = (const float*)d_in[6];   // k
  W_[2] = (const float*)d_in[7];  b_[2] = (const float*)d_in[8];   // vr
  W_[3] = (const float*)d_in[9];  b_[3] = (const float*)d_in[10];  // va
  W_[4] = (const float*)d_in[11]; b_[4] = (const float*)d_in[12];  // or
  W_[5] = (const float*)d_in[13]; b_[5] = (const float*)d_in[14];  // oa
  W_[6] = (const float*)d_in[15]; b_[6] = (const float*)d_in[16];  // sr
  W_[7] = (const float*)d_in[17]; b_[7] = (const float*)d_in[18];  // sa
  W_[8] = (const float*)d_in[19]; b_[8] = (const float*)d_in[20];  // dr
  W_[9] = (const float*)d_in[21]; b_[9] = (const float*)d_in[22];  // da

  // ---- workspace carve-up ----
  char* wp = (char*)d_ws;
  auto alloc = [&](size_t bytes) { char* r = wp; wp += (bytes + 255) & ~(size_t)255; return r; };
  auto allocH = [&](size_t elems) { return (__bf16*)alloc(elems * sizeof(__bf16)); };
  auto allocF = [&](size_t elems) { return (float*)alloc(elems * sizeof(float)); };

  __bf16 *xr_h = allocH(N_RNA * IN_C),  *xr_l = allocH(N_RNA * IN_C);
  __bf16 *xa_h = allocH(N_ATAC * IN_C), *xa_l = allocH(N_ATAC * IN_C);
  __bf16 *Wt_h[10], *Wt_l[10];
  const int Kof[10] = {IN_C, IN_C, IN_C, IN_C, HID, HID, IN_C, IN_C, IN_C, IN_C};
  for (int i = 0; i < 10; ++i) { Wt_h[i] = allocH((size_t)Kof[i] * HID); Wt_l[i] = allocH((size_t)Kof[i] * HID); }
  __bf16 *q_h = allocH(N_RNA * HID),  *q_l = allocH(N_RNA * HID);
  __bf16 *k_h = allocH(N_ATAC * HID), *k_l = allocH(N_ATAC * HID);
  float  *va  = allocF(N_ATAC * HID);
  float  *vr  = allocF(N_RNA * HID);
  __bf16 *sr_h = allocH(N_RNA * HID),  *sr_l = allocH(N_RNA * HID);
  __bf16 *sa_h = allocH(N_ATAC * HID), *sa_l = allocH(N_ATAC * HID);
  __bf16 *rag_h = allocH(N_RNA * HID), *rag_l = allocH(N_RNA * HID);
  float  *aag  = allocF(N_ATAC * HID);
  __bf16 *aag_h = allocH(N_ATAC * HID), *aag_l = allocH(N_ATAC * HID);
  __bf16 *ro_h = allocH(N_RNA * HID),  *ro_l = allocH(N_RNA * HID);
  __bf16 *ao_h = allocH(N_ATAC * HID), *ao_l = allocH(N_ATAC * HID);
  float  *tw = allocF(NH * N_RNA * TOPK);
  int    *ti = (int*)alloc(NH * N_RNA * TOPK * sizeof(int));

  float* out_rna  = (float*)d_out;
  float* out_atac = out_rna + N_RNA * HID;

  // ---- one-time conversions ----
  { int n = N_RNA * IN_C;  fconv_hl<<<(n + 255) / 256, 256, 0, stream>>>(x_rna, xr_h, xr_l, n); }
  { int n = N_ATAC * IN_C; fconv_hl<<<(n + 255) / 256, 256, 0, stream>>>(x_atac, xa_h, xa_l, n); }
  for (int i = 0; i < 10; ++i) {
    int n = Kof[i] * HID;
    wconv_T<<<(n + 255) / 256, 256, 0, stream>>>(W_[i], Wt_h[i], Wt_l[i], Kof[i]);
  }
  { int n = N_ATAC * HID; zerof<<<(n + 255) / 256, 256, 0, stream>>>(aag, n); }

  auto gemm = [&](const __bf16* Xh, const __bf16* Xl, const __bf16* Wh, const __bf16* Wl,
                  const float* bias, float* Yf, __bf16* Yh, __bf16* Yl,
                  int rows, int K, int ldw, int mode, int accum) {
    gemm_bf16hl<<<rows / 16, 256, 0, stream>>>(Xh, Xl, Wh, Wl, bias, Yf, Yh, Yl, K, ldw, mode, accum);
  };

  // ---- input projections [N,256]@[256,128] (bf16-pair outputs except va/vr) ----
  gemm(xr_h, xr_l, Wt_h[0], Wt_l[0], b_[0], nullptr, q_h, q_l, N_RNA,  IN_C, IN_C, 1, 0); // q
  gemm(xa_h, xa_l, Wt_h[1], Wt_l[1], b_[1], nullptr, k_h, k_l, N_ATAC, IN_C, IN_C, 1, 0); // k
  gemm(xr_h, xr_l, Wt_h[2], Wt_l[2], b_[2], vr, nullptr, nullptr, N_RNA,  IN_C, IN_C, 0, 0); // v_r
  gemm(xa_h, xa_l, Wt_h[3], Wt_l[3], b_[3], va, nullptr, nullptr, N_ATAC, IN_C, IN_C, 0, 0); // v_a
  gemm(xr_h, xr_l, Wt_h[6], Wt_l[6], b_[6], nullptr, sr_h, sr_l, N_RNA,  IN_C, IN_C, 1, 0); // s_r
  gemm(xa_h, xa_l, Wt_h[7], Wt_l[7], b_[7], nullptr, sa_h, sa_l, N_ATAC, IN_C, IN_C, 1, 0); // s_a

  // ---- fused attention / top-k ----
  attn_topk<<<NH * (N_RNA / 16), 256, 0, stream>>>(q_h, q_l, k_h, k_l, cmask, tw, ti);

  // ---- sparse aggregations ----
  rna_gather<<<N_RNA, 128, 0, stream>>>(va, tw, ti, rag_h, rag_l);
  atac_scatter<<<NH * N_RNA, 64, 0, stream>>>(vr, tw, ti, aag);
  { int n = N_ATAC * HID; fconv_hl<<<(n + 255) / 256, 256, 0, stream>>>(aag, aag_h, aag_l, n); }

  // ---- head-mix projections [N,128]@[128,128] ----
  gemm(rag_h, rag_l, Wt_h[4], Wt_l[4], b_[4], nullptr, ro_h, ro_l, N_RNA,  HID, HID, 1, 0);
  gemm(aag_h, aag_l, Wt_h[5], Wt_l[5], b_[5], nullptr, ao_h, ao_l, N_ATAC, HID, HID, 1, 0);

  // ---- concat + dim reduction: out = agg@Wd[0:128] + bd + self@Wd[128:256] ----
  gemm(ro_h, ro_l, Wt_h[8],       Wt_l[8],       b_[8],   out_rna,  nullptr, nullptr, N_RNA,  HID, IN_C, 0, 0);
  gemm(sr_h, sr_l, Wt_h[8] + HID, Wt_l[8] + HID, nullptr, out_rna,  nullptr, nullptr, N_RNA,  HID, IN_C, 0, 1);
  gemm(ao_h, ao_l, Wt_h[9],       Wt_l[9],       b_[9],   out_atac, nullptr, nullptr, N_ATAC, HID, IN_C, 0, 0);
  gemm(sa_h, sa_l, Wt_h[9] + HID, Wt_l[9] + HID, nullptr, out_atac, nullptr, nullptr, N_ATAC, HID, IN_C, 0, 1);
}
